// MixvMFGrad_14542759264366
// MI455X (gfx1250) — compile-verified, matrix operand order checked by reading, not run on hardware
//
#include <hip/hip_runtime.h>
#include <hip/hip_bf16.h>

typedef _Float16 v16h __attribute__((ext_vector_type(16)));
typedef _Float16 v8h  __attribute__((ext_vector_type(8)));
typedef float    v8f  __attribute__((ext_vector_type(8)));
typedef float    v4f  __attribute__((ext_vector_type(4)));

#define D_DIM     512
#define K_COMP    64
#define WAVES     8
#define ROWS_PB   128      // 8 waves * 16 rows
#define MUSA_S    520      // padded f16 stride for mus row-major (bank-conflict-free)
#define MUST_S    72       // padded f16 stride for mus transposed
#define DOTS_S    65       // padded f32 stride for per-wave dots scratch
#define W_S       72       // padded f16 stride for per-wave weights

// log of vMF normalizer: d*(-0.5 log 2pi) + nu*log(kappa) - log I_nu(kappa),
// with nu = d/2-1 = 255, via uniform asymptotic expansion (A&S 9.7.7).
__device__ __forceinline__ float logc_vmf(float kappa) {
  const float nu = 255.0f;
  float z  = kappa * (1.0f / nu);
  float z2 = z * z;
  float sq = sqrtf(1.0f + z2);
  float eta = sq + logf(z) - log1pf(sq);
  float t  = 1.0f / sq;
  float t2 = t * t;
  float u1 = (3.0f * t - 5.0f * t * t2) * (1.0f / 24.0f);
  float u2 = (81.0f * t2 - 462.0f * t2 * t2 + 385.0f * t2 * t2 * t2) * (1.0f / 1152.0f);
  float lI = nu * eta
           - 3.68957031f                 // 0.5*log(2*pi*255)
           - 0.25f * log1pf(z2)
           + log1pf(u1 * (1.0f / nu) + u2 * (1.0f / (nu * nu)));
  return -470.49652900f                  // 512 * (-0.5*log(2*pi))
       + nu * logf(kappa) - lI;
}

// Combine two 8-halves LDS loads into one 16-elem fragment (ds_load_b128 x2).
__device__ __forceinline__ v16h frag16(const _Float16* p0) {
  v8h x = *(const v8h*)p0;
  v8h y = *(const v8h*)(p0 + 16);
  return __builtin_shufflevector(x, y, 0,1,2,3,4,5,6,7,8,9,10,11,12,13,14,15);
}

__global__ __launch_bounds__(256) void vmf_mix_grad_kernel(
    const float* __restrict__ s, const float* __restrict__ alphas,
    const float* __restrict__ mus, const float* __restrict__ kappas,
    float* __restrict__ out, int bs)
{
  __shared__ _Float16 musA[K_COMP * MUSA_S];   // mus[comp][k]  (B of GEMM1)
  __shared__ _Float16 musT[D_DIM * MUST_S];    // mus[k? no: musT[j][comp]] (B of GEMM2)
  __shared__ float    dotsS[WAVES][16 * DOTS_S];
  __shared__ _Float16 wS[WAVES][16 * W_S];
  __shared__ float    gsS[WAVES][16];
  __shared__ float    ckS[K_COMP], kapS[K_COMP], rkapS[K_COMP];

  const int tid  = threadIdx.x;
  const int wv   = tid >> 5;
  const int lane = tid & 31;
  const int n    = lane & 15;   // column / row-within-tile index (ISA frag layouts)
  const int hk   = lane >> 4;   // half-select in fragment layouts

  // ---- stage mus (f32 -> f16) into both LDS layouts: 64x512 = 8192 float4 ----
  #pragma unroll 4
  for (int it = 0; it < 32; ++it) {
    int idx  = tid + it * 256;
    int comp = idx >> 7;                 // / (512/4)
    int kp   = (idx & 127) << 2;
    v4f v = *(const v4f*)(mus + comp * D_DIM + kp);
    #pragma unroll
    for (int i = 0; i < 4; ++i) {
      _Float16 h = (_Float16)v[i];
      musA[comp * MUSA_S + kp + i] = h;
      musT[(kp + i) * MUST_S + comp] = h;
    }
  }
  if (tid < K_COMP) {
    float kap = kappas[tid];
    kapS[tid]  = kap;
    rkapS[tid] = 1.0f / kap;
    ckS[tid]   = logf(alphas[tid]) + logf(kap) + logc_vmf(kap);
  }
  __syncthreads();

  const int base = blockIdx.x * ROWS_PB + wv * 16;

  // =========== GEMM1: dots[16 x 64] = s_tile[16 x 512] * mus^T ===========
  const int rowA = min(base + n, bs - 1);            // A-fragment row (m = lane&15)
  const float* srow = s + (size_t)rowA * D_DIM;

  v8f acc[4];
  #pragma unroll
  for (int t = 0; t < 4; ++t) acc[t] = v8f{0.f,0.f,0.f,0.f,0.f,0.f,0.f,0.f};

  for (int kb = 0; kb < 16; ++kb) {
    const int k0 = kb * 32;
    // A fragment: a[0..7]=s[m][k0+hk*8 ..], a[8..15]=s[m][k0+16+hk*8 ..]
    v4f f0 = *(const v4f*)(srow + k0 + hk * 8);
    v4f f1 = *(const v4f*)(srow + k0 + hk * 8 + 4);
    v4f f2 = *(const v4f*)(srow + k0 + 16 + hk * 8);
    v4f f3 = *(const v4f*)(srow + k0 + 16 + hk * 8 + 4);
    v16h a;
    #pragma unroll
    for (int i = 0; i < 4; ++i) {
      a[i]      = (_Float16)f0[i];
      a[4 + i]  = (_Float16)f1[i];
      a[8 + i]  = (_Float16)f2[i];
      a[12 + i] = (_Float16)f3[i];
    }
    #pragma unroll
    for (int t = 0; t < 4; ++t) {     // component tiles n0 = t*16
      v16h b = frag16(&musA[(t * 16 + n) * MUSA_S + k0 + hk * 8]);
      acc[t] = __builtin_amdgcn_wmma_f32_16x16x32_f16(
                 false, a, false, b, (short)0, acc[t], false, false);
    }
  }

  // epilogue: val = ck + kappa*dot -> per-wave LDS scratch (D layout: m=e+8*hk)
  {
    float ckr[4], kpr[4];
    #pragma unroll
    for (int t = 0; t < 4; ++t) { ckr[t] = ckS[t * 16 + n]; kpr[t] = kapS[t * 16 + n]; }
    float* dw = dotsS[wv];
    #pragma unroll
    for (int t = 0; t < 4; ++t)
      #pragma unroll
      for (int e = 0; e < 8; ++e)
        dw[(e + 8 * hk) * DOTS_S + t * 16 + n] = ckr[t] + kpr[t] * acc[t][e];
  }
  __syncthreads();

  // =========== row softmax weights + g.s = sum_c w_c * dot_c ===========
  {
    float* dw = dotsS[wv];
    _Float16* ww = wS[wv];
    if (lane < 16) {
      const int r = lane;
      float mx = -3.4e38f;
      for (int c = 0; c < K_COMP; ++c) mx = fmaxf(mx, dw[r * DOTS_S + c]);
      float gs = 0.f;
      for (int c = 0; c < K_COMP; ++c) {
        float val = dw[r * DOTS_S + c];
        float w = __expf(val - mx);
        ww[r * W_S + c] = (_Float16)w;
        gs += w * ((val - ckS[c]) * rkapS[c]);   // recover dot, accumulate g.s
      }
      gsS[wv][r] = gs;
    }
  }
  __syncthreads();

  // A fragments of GEMM2: w[16 x 64] (two k-chunks of 32)
  v16h wA[2];
  #pragma unroll
  for (int cb = 0; cb < 2; ++cb)
    wA[cb] = frag16(&wS[wv][n * W_S + cb * 32 + hk * 8]);

  // =========== GEMM2 pass 1: accumulate ||g||^2 per row ===========
  float nrm[8];
  #pragma unroll
  for (int e = 0; e < 8; ++e) nrm[e] = 0.f;

  for (int jt = 0; jt < 32; ++jt) {
    v8f d = v8f{0.f,0.f,0.f,0.f,0.f,0.f,0.f,0.f};
    #pragma unroll
    for (int cb = 0; cb < 2; ++cb) {
      v16h b = frag16(&musT[(jt * 16 + n) * MUST_S + cb * 32 + hk * 8]);
      d = __builtin_amdgcn_wmma_f32_16x16x32_f16(
            false, wA[cb], false, b, (short)0, d, false, false);
    }
    #pragma unroll
    for (int e = 0; e < 8; ++e) nrm[e] += d[e] * d[e];
  }
  // reduce across the 16 column-lanes (rows live in (e, hk) only)
  #pragma unroll
  for (int e = 0; e < 8; ++e) {
    float v = nrm[e];
    v += __shfl_xor(v, 1, 32);
    v += __shfl_xor(v, 2, 32);
    v += __shfl_xor(v, 4, 32);
    v += __shfl_xor(v, 8, 32);
    nrm[e] = rsqrtf(fmaxf(v, 1e-30f));
  }
  float gsv[8];
  #pragma unroll
  for (int e = 0; e < 8; ++e) gsv[e] = gsS[wv][e + 8 * hk];

  // =========== GEMM2 pass 2: out = (g - s*(g.s)) * rsqrt(||g||^2) ===========
  for (int jt = 0; jt < 32; ++jt) {
    v8f d = v8f{0.f,0.f,0.f,0.f,0.f,0.f,0.f,0.f};
    #pragma unroll
    for (int cb = 0; cb < 2; ++cb) {
      v16h b = frag16(&musT[(jt * 16 + n) * MUST_S + cb * 32 + hk * 8]);
      d = __builtin_amdgcn_wmma_f32_16x16x32_f16(
            false, wA[cb], false, b, (short)0, d, false, false);
    }
    const int j = jt * 16 + n;
    #pragma unroll
    for (int e = 0; e < 8; ++e) {
      int row = base + e + 8 * hk;
      int rcl = min(row, bs - 1);
      float sv = s[(size_t)rcl * D_DIM + j];
      float o  = (d[e] - sv * gsv[e]) * nrm[e];
      if (row < bs) out[(size_t)row * D_DIM + j] = o;
    }
  }
}

extern "C" void kernel_launch(void* const* d_in, const int* in_sizes, int n_in,
                              void* d_out, int out_size, void* d_ws, size_t ws_size,
                              hipStream_t stream) {
  (void)n_in; (void)out_size; (void)d_ws; (void)ws_size;
  const float* s      = (const float*)d_in[0];
  const float* alphas = (const float*)d_in[1];
  const float* mus    = (const float*)d_in[2];
  const float* kappas = (const float*)d_in[3];
  float* out = (float*)d_out;
  const int bs = in_sizes[0] / D_DIM;
  const int blocks = (bs + ROWS_PB - 1) / ROWS_PB;
  hipLaunchKernelGGL(vmf_mix_grad_kernel, dim3(blocks), dim3(256), 0, stream,
                     s, alphas, mus, kappas, out, bs);
}
